// ActionPrototypeLayer_40312563041000
// MI455X (gfx1250) — matrix operand order repaired and checked by reading, not compiled, not run on hardware
//
#include <hip/hip_runtime.h>
#include <hip/hip_bf16.h>

typedef __attribute__((ext_vector_type(16))) __bf16 v16bf;
typedef __attribute__((ext_vector_type(8)))  __bf16 v8bf;
typedef __attribute__((ext_vector_type(8)))  float  v8f;

#define B_ROWS 16384
#define DHID   1024
#define NA     64
#define NP     32
#define NCOL   2048   // NA*NP

#define TM 128
#define TN 128
#define TK 32
#define KSTEPS (DHID / TK)   // 32

// CDNA5 async global->LDS path (ASYNCcnt) if the toolchain exposes it.
#if __has_builtin(__builtin_amdgcn_global_load_async_to_lds_b128) && \
    __has_builtin(__builtin_amdgcn_s_wait_asynccnt)
#define USE_ASYNC_LDS 1
#else
#define USE_ASYNC_LDS 0
#endif

#if USE_ASYNC_LDS
#define GAS __attribute__((address_space(1)))
#define LAS __attribute__((address_space(3)))
typedef int v4i __attribute__((vector_size(16)));   // builtin's pointee type
// copy 32 bytes (16 bf16) global -> LDS per lane, as two b128 async ops;
// INST_OFFSET applies to both the LDS and global addresses (ISA 10.x), so the
// same base pointers with ioffset 16 cover the second 16-byte chunk.
__device__ __forceinline__ void async_cp32(const unsigned short* g,
                                           unsigned short* l) {
  v4i* gg = (v4i*)(void*)g;                 // drop const, retype (generic AS)
  v4i* ll = (v4i*)(void*)l;
  GAS v4i* gp = (GAS v4i*)gg;               // addrspacecast generic -> global
  LAS v4i* lp = (LAS v4i*)ll;               // addrspacecast generic -> LDS
  __builtin_amdgcn_global_load_async_to_lds_b128(gp, lp, 0, 0);
  __builtin_amdgcn_global_load_async_to_lds_b128(gp, lp, 16, 0);
}
#endif

// ---------------------------------------------------------------------------
// K0: zero accumulators (must re-init every call; harness poisons ws once)
// ---------------------------------------------------------------------------
__global__ void k_init(float* __restrict__ total, float* __restrict__ pos,
                       float* __restrict__ loss) {
  int i = blockIdx.x * blockDim.x + threadIdx.x;
  if (i < B_ROWS) { total[i] = 0.0f; pos[i] = 0.0f; }
  if (i == 0) loss[0] = 0.0f;
}

// ---------------------------------------------------------------------------
// K1/K2: per-row L2 normalize (fp32 -> bf16), save norm.
// One 256-thread block per row of 1024; each thread handles 4 elements.
// ---------------------------------------------------------------------------
__global__ __launch_bounds__(256)
void k_rownorm(const float* __restrict__ x, unsigned short* __restrict__ y,
               float* __restrict__ norms) {
  const int row  = blockIdx.x;
  const int tid  = threadIdx.x;
  const int lane = tid & 31, wv = tid >> 5;
  __shared__ float sred[8];
  __shared__ float s_scale;

  const float* xr = x + (size_t)row * DHID;
  float4 v = *(const float4*)(xr + tid * 4);
  float ss = v.x * v.x + v.y * v.y + v.z * v.z + v.w * v.w;
#pragma unroll
  for (int off = 16; off > 0; off >>= 1) ss += __shfl_xor(ss, off, 32);
  if (lane == 0) sred[wv] = ss;
  __syncthreads();
  if (tid == 0) {
    float t = 0.0f;
#pragma unroll
    for (int i = 0; i < 8; ++i) t += sred[i];
    float n = sqrtf(t);
    norms[row] = n;
    s_scale = 1.0f / fmaxf(n, 1e-12f);   // F.normalize eps
  }
  __syncthreads();
  const float sc = s_scale;
  __bf16 h0 = (__bf16)(v.x * sc), h1 = (__bf16)(v.y * sc);
  __bf16 h2 = (__bf16)(v.z * sc), h3 = (__bf16)(v.w * sc);
  ushort4 o;
  o.x = __builtin_bit_cast(unsigned short, h0);
  o.y = __builtin_bit_cast(unsigned short, h1);
  o.z = __builtin_bit_cast(unsigned short, h2);
  o.w = __builtin_bit_cast(unsigned short, h3);
  *(ushort4*)(y + (size_t)row * DHID + tid * 4) = o;
}

// ---------------------------------------------------------------------------
// K3: fused bf16 WMMA GEMM  sim = np_pool @ np_prot^T  (M=16384,N=2048,K=1024)
//     + exp(sim/T) reduction into total[b], pos[b] + attention-logit extract.
// Block tile 128x128, 8 waves of 64x32, K-step 32, double-buffered LDS with
// CDNA5 async global->LDS staging (fallback: VGPR staging).
// ---------------------------------------------------------------------------
__global__ __launch_bounds__(256, 2)
void k_sim_wmma(const unsigned short* __restrict__ np_pool,
                const unsigned short* __restrict__ np_prot,
                const float* __restrict__ pool_norm,
                const float* __restrict__ prot_norm,
                const int* __restrict__ app_ids,
                float* __restrict__ total, float* __restrict__ pos,
                float* __restrict__ scores) {
  __shared__ __align__(32) unsigned short sA[2][TM * TK];
  __shared__ __align__(32) unsigned short sB[2][TN * TK];
  __shared__ float s_total[TM], s_pos[TM], s_pooln[TM], s_protn[TN];
  __shared__ int   s_app[TM];

  const int tid  = threadIdx.x;
  const int lane = tid & 31, wv = tid >> 5;     // wave32
  const int lm   = lane & 15, lhi = lane >> 4;
  const int wm   = (wv & 1) * 64;               // wave M offset in block tile
  const int wn   = (wv >> 1) * 32;              // wave N offset in block tile
  const int m0   = blockIdx.y * TM;
  const int n0   = blockIdx.x * TN;

  if (tid < TM) {
    s_total[tid] = 0.0f;
    s_pos[tid]   = 0.0f;
    s_app[tid]   = app_ids[m0 + tid];
    s_pooln[tid] = pool_norm[m0 + tid];
    s_protn[tid] = prot_norm[n0 + tid];
  }

  // each thread stages 16 bf16 (32B) of A and of B per K-step
  const int lrow = tid >> 1;
  const int lkc  = (tid & 1) * 16;
  const size_t aBase = (size_t)(m0 + lrow) * DHID + lkc;
  const size_t bBase = (size_t)(n0 + lrow) * DHID + lkc;
  const int    lidx  = lrow * TK + lkc;

#if USE_ASYNC_LDS
  async_cp32(np_pool + aBase, &sA[0][lidx]);
  async_cp32(np_prot + bBase, &sB[0][lidx]);
  __builtin_amdgcn_s_wait_asynccnt(0);
  __syncthreads();
#else
  uint4 ra0 = *(const uint4*)(np_pool + aBase);
  uint4 ra1 = *(const uint4*)(np_pool + aBase + 8);
  uint4 rb0 = *(const uint4*)(np_prot + bBase);
  uint4 rb1 = *(const uint4*)(np_prot + bBase + 8);
  *(uint4*)&sA[0][lidx]     = ra0;
  *(uint4*)&sA[0][lidx + 8] = ra1;
  *(uint4*)&sB[0][lidx]     = rb0;
  *(uint4*)&sB[0][lidx + 8] = rb1;
  __syncthreads();
#endif

  v8f acc[4][2] = {};
  int buf = 0;
  for (int kt = 0; kt < KSTEPS; ++kt) {
#if USE_ASYNC_LDS
    if (kt + 1 < KSTEPS) {               // DMA next K tile into back buffer
      const size_t ko = (size_t)(kt + 1) * TK;
      const int nb = buf ^ 1;
      async_cp32(np_pool + aBase + ko, &sA[nb][lidx]);
      async_cp32(np_prot + bBase + ko, &sB[nb][lidx]);
    }
#else
    if (kt + 1 < KSTEPS) {               // register-prefetch next K tile
      const size_t ko = (size_t)(kt + 1) * TK;
      ra0 = *(const uint4*)(np_pool + aBase + ko);
      ra1 = *(const uint4*)(np_pool + aBase + ko + 8);
      rb0 = *(const uint4*)(np_prot + bBase + ko);
      rb1 = *(const uint4*)(np_prot + bBase + ko + 8);
    }
#endif
    // A fragments: code's own contiguous k order — lane(hi) selects K half,
    // elems 0..15 -> K = lhi*16 + j (contiguous, matches B)
    v16bf afr[4];
#pragma unroll
    for (int mi = 0; mi < 4; ++mi) {
      const unsigned short* base = &sA[buf][(wm + mi * 16 + lm) * TK];
      v8bf lo = *(const v8bf*)(base + lhi * 16);
      v8bf hi = *(const v8bf*)(base + lhi * 16 + 8);
      afr[mi] = __builtin_shufflevector(lo, hi,
                0,1,2,3,4,5,6,7,8,9,10,11,12,13,14,15);
    }
    // B fragments: same contiguous k order as A — lane = (khi<<4)|n, elems = K run
    v16bf bfr[2];
#pragma unroll
    for (int nj = 0; nj < 2; ++nj)
      bfr[nj] = *(const v16bf*)&sB[buf][(wn + nj * 16 + lm) * TK + lhi * 16];

#pragma unroll
    for (int mi = 0; mi < 4; ++mi)
#pragma unroll
      for (int nj = 0; nj < 2; ++nj)
        acc[mi][nj] = __builtin_amdgcn_wmma_f32_16x16x32_bf16(
            false, afr[mi], false, bfr[nj], (short)0, acc[mi][nj],
            false, false);

    if (kt + 1 < KSTEPS) {
#if USE_ASYNC_LDS
      __builtin_amdgcn_s_wait_asynccnt(0);
      __syncthreads();
      buf ^= 1;
#else
      const int nb = buf ^ 1;
      *(uint4*)&sA[nb][lidx]     = ra0;
      *(uint4*)&sA[nb][lidx + 8] = ra1;
      *(uint4*)&sB[nb][lidx]     = rb0;
      *(uint4*)&sB[nb][lidx + 8] = rb1;
      __syncthreads();
      buf = nb;
#endif
    }
  }

  // epilogue: C/D layout — lane(&15)=N, lane(>>4)*8 + vgpr = M
  const float invT = 1.0f / 0.07f;
#pragma unroll
  for (int mi = 0; mi < 4; ++mi) {
#pragma unroll
    for (int nj = 0; nj < 2; ++nj) {
#pragma unroll
      for (int r = 0; r < 8; ++r) {
        const int ml = wm + mi * 16 + lhi * 8 + r;
        const int nl = wn + nj * 16 + lm;
        const float dotn = acc[mi][nj][r];
        const float e = __expf(dotn * invT);
        atomicAdd(&s_total[ml], e);                    // ds_add_f32
        const int n = n0 + nl;
        if ((n >> 5) == s_app[ml]) {
          atomicAdd(&s_pos[ml], e);
          // raw score = dotnorm * |pool| * |proto| / sqrt(1024)
          scores[(size_t)(m0 + ml) * NP + (n & 31)] =
              dotn * s_pooln[ml] * s_protn[nl] * 0.03125f;
        }
      }
    }
  }
  __syncthreads();
  if (tid < TM) {
    atomicAdd(&total[m0 + tid], s_total[tid]);
    atomicAdd(&pos[m0 + tid],   s_pos[tid]);
  }
}

// ---------------------------------------------------------------------------
// K4: softmax over 32 logits, weighted prototype sum, residual + LayerNorm,
//     per-sample contrastive loss accumulation. One block per batch row.
// ---------------------------------------------------------------------------
__global__ __launch_bounds__(256)
void k_attn_ln(const float* __restrict__ pooled,
               const int* __restrict__ app_ids,
               const float* __restrict__ protos,
               const float* __restrict__ gamma,
               const float* __restrict__ beta,
               const float* __restrict__ scores,
               const float* __restrict__ total,
               const float* __restrict__ pos,
               float* __restrict__ out, float* __restrict__ loss_acc) {
  const int b = blockIdx.x, tid = threadIdx.x;
  const int lane = tid & 31, wv = tid >> 5;
  __shared__ float s_attn[NP];
  __shared__ float sred0[8], sred1[8];
  __shared__ float s_mu, s_rstd;

  const int a = app_ids[b];
  if (tid < 32) {
    float s = scores[(size_t)b * NP + tid];
    float mx = s;
#pragma unroll
    for (int off = 16; off > 0; off >>= 1) mx = fmaxf(mx, __shfl_xor(mx, off, 32));
    float e = __expf(s - mx);
    float sm = e;
#pragma unroll
    for (int off = 16; off > 0; off >>= 1) sm += __shfl_xor(sm, off, 32);
    s_attn[tid] = e / sm;
  }
  __syncthreads();

  const int d0 = tid * 4;
  const float* pr = protos + (size_t)a * NP * DHID + d0;
  float a0 = 0.f, a1 = 0.f, a2 = 0.f, a3 = 0.f;
#pragma unroll 8
  for (int p = 0; p < NP; ++p) {
    const float w = s_attn[p];
    float4 v = *(const float4*)(pr + (size_t)p * DHID);
    a0 += w * v.x; a1 += w * v.y; a2 += w * v.z; a3 += w * v.w;
  }
  float4 po = *(const float4*)(pooled + (size_t)b * DHID + d0);
  const float r0 = a0 + po.x, r1 = a1 + po.y, r2 = a2 + po.z, r3 = a3 + po.w;

  float lsum = r0 + r1 + r2 + r3;
  float lsq  = r0 * r0 + r1 * r1 + r2 * r2 + r3 * r3;
#pragma unroll
  for (int off = 16; off > 0; off >>= 1) {
    lsum += __shfl_xor(lsum, off, 32);
    lsq  += __shfl_xor(lsq,  off, 32);
  }
  if (lane == 0) { sred0[wv] = lsum; sred1[wv] = lsq; }
  __syncthreads();
  if (tid == 0) {
    float ts = 0.f, tq = 0.f;
#pragma unroll
    for (int i = 0; i < 8; ++i) { ts += sred0[i]; tq += sred1[i]; }
    const float mu  = ts * (1.0f / DHID);
    const float var = tq * (1.0f / DHID) - mu * mu;
    s_mu = mu;
    s_rstd = rsqrtf(fmaxf(var, 0.0f) + 1e-5f);
  }
  __syncthreads();
  const float mu = s_mu, rstd = s_rstd;
  float4 g = *(const float4*)(gamma + d0);
  float4 bt = *(const float4*)(beta + d0);
  float4 o;
  o.x = (r0 - mu) * rstd * g.x + bt.x;
  o.y = (r1 - mu) * rstd * g.y + bt.y;
  o.z = (r2 - mu) * rstd * g.z + bt.z;
  o.w = (r3 - mu) * rstd * g.w + bt.w;
  *(float4*)(out + (size_t)b * DHID + d0) = o;

  if (tid == 0) {
    const float l = -logf(pos[b] / (total[b] + 1e-10f));
    atomicAdd(loss_acc, l * (1.0f / (float)B_ROWS));
  }
}

// ---------------------------------------------------------------------------
// K5: loss = max(mean, 0)
// ---------------------------------------------------------------------------
__global__ void k_finalize(const float* __restrict__ loss_acc,
                           float* __restrict__ out) {
  out[(size_t)B_ROWS * DHID] = fmaxf(loss_acc[0], 0.0f);
}

// ---------------------------------------------------------------------------
extern "C" void kernel_launch(void* const* d_in, const int* in_sizes, int n_in,
                              void* d_out, int out_size, void* d_ws, size_t ws_size,
                              hipStream_t stream) {
  (void)in_sizes; (void)n_in; (void)out_size; (void)ws_size;
  const float* pooled  = (const float*)d_in[0];
  const int*   app_ids = (const int*)d_in[1];
  const float* protos  = (const float*)d_in[2];
  const float* gamma   = (const float*)d_in[3];
  const float* beta    = (const float*)d_in[4];
  float* out = (float*)d_out;

  char* ws = (char*)d_ws;
  size_t off = 0;
  auto alloc = [&](size_t bytes) -> char* {
    char* p = ws + off;
    off += (bytes + 255) & ~(size_t)255;
    return p;
  };
  unsigned short* np_pool = (unsigned short*)alloc((size_t)B_ROWS * DHID * 2);
  unsigned short* np_prot = (unsigned short*)alloc((size_t)NCOL * DHID * 2);
  float* pool_norm = (float*)alloc((size_t)B_ROWS * 4);
  float* prot_norm = (float*)alloc((size_t)NCOL * 4);
  float* total     = (float*)alloc((size_t)B_ROWS * 4);
  float* pos       = (float*)alloc((size_t)B_ROWS * 4);
  float* scores    = (float*)alloc((size_t)B_ROWS * NP * 4);
  float* loss_acc  = (float*)alloc(256);

  k_init<<<(B_ROWS + 255) / 256, 256, 0, stream>>>(total, pos, loss_acc);
  k_rownorm<<<NCOL,   256, 0, stream>>>(protos, np_prot, prot_norm);
  k_rownorm<<<B_ROWS, 256, 0, stream>>>(pooled, np_pool, pool_norm);
  k_sim_wmma<<<dim3(NCOL / TN, B_ROWS / TM), 256, 0, stream>>>(
      np_pool, np_prot, pool_norm, prot_norm, app_ids, total, pos, scores);
  k_attn_ln<<<B_ROWS, 256, 0, stream>>>(pooled, app_ids, protos, gamma, beta,
                                        scores, total, pos, out, loss_acc);
  k_finalize<<<1, 1, 0, stream>>>(loss_acc, out);
}
